// RNNTJoint_83348135346397
// MI455X (gfx1250) — matrix operand order repaired and checked by reading, not compiled, not run on hardware
//
#include <hip/hip_runtime.h>
#include <hip/hip_bf16.h>
#include <stdint.h>

// ---------------- problem constants ----------------
#define BB 4
#define TT 256
#define UU 128
#define EE 1024
#define PP 640
#define HH 640
#define VV 1025
#define VP 1024          // WMMA-covered columns (col 1024 handled by VALU)
#define HP 648           // padded bf16 row stride in LDS (16B aligned, bank-friendly)

typedef float v8f   __attribute__((ext_vector_type(8)));
typedef __bf16 v16bf __attribute__((ext_vector_type(16)));
typedef unsigned int uint32x4 __attribute__((ext_vector_type(4)));
typedef int int32x4 __attribute__((ext_vector_type(4)));
typedef int int32x8 __attribute__((ext_vector_type(8)));

union FragB { uint4 u[2]; v16bf v; };

__device__ __forceinline__ unsigned short f2bf(float f) {
  unsigned int x = __float_as_uint(f);
  unsigned int r = x + 0x7fffu + ((x >> 16) & 1u);   // round-to-nearest-even
  return (unsigned short)(r >> 16);
}
__device__ __forceinline__ float bf2f(unsigned short u) {
  return __uint_as_float(((unsigned int)u) << 16);
}

// ---- CDNA5 async global->LDS copy (ASYNCcnt) ----
__device__ __forceinline__ void async_copy_b128(unsigned int lds_off, const void* gptr) {
  asm volatile("global_load_async_to_lds_b128 %0, %1, off"
               :: "v"(lds_off), "v"((unsigned long long)(uintptr_t)gptr)
               : "memory");
}
__device__ __forceinline__ void wait_async0() {
  asm volatile("s_wait_asynccnt 0x0" ::: "memory");
}

// ---- CDNA5 Tensor Data Mover: 2-D tile (data_size=4B) global -> LDS ----
#if __has_builtin(__builtin_amdgcn_tensor_load_to_lds)
#define HAVE_TDM 1
__device__ __forceinline__ void tdm_load_2d_f32(unsigned int lds_addr, const void* gptr,
                                                unsigned int tile_d0, unsigned int tile_d1,
                                                unsigned int tensor_d0, unsigned int stride0) {
  unsigned long long ga = (unsigned long long)(uintptr_t)gptr;
  uint32x4 g0;
  g0[0] = 1u;                                   // count=1 valid descriptor
  g0[1] = lds_addr;                             // lds_addr [63:32]
  g0[2] = (unsigned int)ga;                     // global_addr[31:0]
  g0[3] = (unsigned int)((ga >> 32) & 0x01ffffffu) | 0x80000000u; // addr[56:32] | type=2
  int32x8 g1;
  g1[0] = (int)(2u << 16);                      // data_size=2 (4 bytes), no multicast
  g1[1] = (int)((tensor_d0 & 0xffffu) << 16);   // tensor_dim0 lo16 at bits 63:48
  g1[2] = (int)(((tensor_d0 >> 16) & 0xffffu) | ((tile_d1 & 0xffffu) << 16)); // d0 hi | tensor_dim1 lo
  g1[3] = (int)(((tile_d1 >> 16) & 0xffffu) | ((tile_d0 & 0xffffu) << 16));   // d1 hi | tile_dim0
  g1[4] = (int)(tile_d1 & 0xffffu);             // tile_dim1 | tile_dim2=0
  g1[5] = (int)stride0;                         // tensor_dim0_stride lo32
  g1[6] = 0;
  g1[7] = 0;
  int32x4 gz4 = {0, 0, 0, 0};
#if defined(__clang_major__) && (__clang_major__ >= 23)
  int32x8 gz8 = {0, 0, 0, 0, 0, 0, 0, 0};
  __builtin_amdgcn_tensor_load_to_lds(g0, g1, gz4, gz4, gz8, 0);
#else
  __builtin_amdgcn_tensor_load_to_lds(g0, g1, gz4, gz4, 0);
#endif
}
#else
#define HAVE_TDM 0
#endif

// ---------------- pack: (Bn, C, L) fp32 -> (Bn*L, C) bf16 row-major ----------------
__global__ void k_transpose_cast(const float* __restrict__ src,
                                 unsigned short* __restrict__ dst,
                                 int Bn, int C, int L) {
  int i = blockIdx.x * blockDim.x + threadIdx.x;
  int total = Bn * C * L;
  if (i >= total) return;
  int c = i % C;
  int rest = i / C;
  int l = rest % L;
  int b = rest / L;
  dst[i] = f2bf(src[((long)b * C + c) * L + l]);
}

// ---------------- pack: W (N,K) fp32 row-major -> bf16 WMMA B-fragment layout -------
__global__ void k_packB(const float* __restrict__ W,
                        unsigned short* __restrict__ dst, int N, int K) {
  int i = blockIdx.x * blockDim.x + threadIdx.x;
  if (i >= N * K) return;
  int j    = i & 15;
  int lane = (i >> 4) & 31;
  int rest = i >> 9;
  int NT = N >> 4;
  int nt = rest % NT;
  int kc = rest / NT;
  int n = nt * 16 + (lane & 15);
  int k = kc * 32 + ((j >> 3) << 4) + ((lane >> 4) << 3) + (j & 7);
  dst[i] = f2bf(W[(long)n * K + k]);
}

// ---------------- pack: W_joint row 1024 -> bf16[640] ----------------
__global__ void k_pack_w1024(const float* __restrict__ Wj, unsigned short* __restrict__ dst) {
  int i = blockIdx.x * blockDim.x + threadIdx.x;
  if (i < HH) dst[i] = f2bf(Wj[(long)1024 * HH + i]);
}

// ---------------- generic bf16 WMMA GEMM: C(M,N) = A(M,K)*B + bias ----------------
__global__ __launch_bounds__(32)
void k_gemm_bias(const unsigned short* __restrict__ A,   // (M,K) bf16 row-major
                 const unsigned short* __restrict__ Bp,  // packed fragments
                 const float* __restrict__ bias,         // (N)
                 float* __restrict__ C,                   // (M,N) fp32
                 int M, int N, int K) {
  int lane = threadIdx.x;
  int m0 = blockIdx.x * 16;
  int n0 = blockIdx.y * 64;
  int NT = N >> 4;
  int KC = K >> 5;
  int row = m0 + (lane & 15);
  int kb  = (lane >> 4) << 3;

  v8f acc[4];
  #pragma unroll
  for (int t = 0; t < 4; ++t)
    #pragma unroll
    for (int r = 0; r < 8; ++r) acc[t][r] = 0.f;

  const unsigned short* arow = A + (long)row * K;
  for (int kc = 0; kc < KC; ++kc) {
    FragB a, b[4];
    a.u[0] = *(const uint4*)(arow + kc * 32 + kb);
    a.u[1] = *(const uint4*)(arow + kc * 32 + kb + 16);
    #pragma unroll
    for (int t = 0; t < 4; ++t) {
      const unsigned short* bp = Bp + (((long)kc * NT + (n0 >> 4) + t) * 32 + lane) * 16;
      b[t].u[0] = ((const uint4*)bp)[0];
      b[t].u[1] = ((const uint4*)bp)[1];
    }
    #pragma unroll
    for (int t = 0; t < 4; ++t)
      acc[t] = __builtin_amdgcn_wmma_f32_16x16x32_bf16(
          false, a.v, false, b[t].v, (short)0, acc[t], false, false);
  }
  int mrow = m0 + ((lane >> 4) << 3);
  #pragma unroll
  for (int t = 0; t < 4; ++t) {
    int n = n0 + t * 16 + (lane & 15);
    float bv = bias[n];
    #pragma unroll
    for (int r = 0; r < 8; ++r)
      C[(long)(mrow + r) * N + n] = acc[t][r] + bv;
  }
}

// ---- k_joint helpers: B-tile register loads + one K-step of 16 WMMAs ----
__device__ __forceinline__ void load_btiles(const unsigned short* __restrict__ Wp,
                                            int kc, int w, int lane, FragB bb[8]) {
  #pragma unroll
  for (int i = 0; i < 8; ++i) {
    const unsigned short* bp = Wp + (((long)kc * 64 + w * 8 + i) * 32 + lane) * 16;
    bb[i].u[0] = ((const uint4*)bp)[0];
    bb[i].u[1] = ((const uint4*)bp)[1];
  }
}
__device__ __forceinline__ void wmma_step(const unsigned short* hr0,
                                          const unsigned short* hr1,
                                          int kc, int kb, FragB bb[8], v8f acc[2][8]) {
  FragB a0, a1;
  a0.u[0] = *(const uint4*)(hr0 + kc * 32 + kb);
  a0.u[1] = *(const uint4*)(hr0 + kc * 32 + kb + 16);
  a1.u[0] = *(const uint4*)(hr1 + kc * 32 + kb);
  a1.u[1] = *(const uint4*)(hr1 + kc * 32 + kb + 16);
  #pragma unroll
  for (int i = 0; i < 8; ++i) {
    acc[0][i] = __builtin_amdgcn_wmma_f32_16x16x32_bf16(
        false, a0.v, false, bb[i].v, (short)0, acc[0][i], false, false);
    acc[1][i] = __builtin_amdgcn_wmma_f32_16x16x32_bf16(
        false, a1.v, false, bb[i].v, (short)0, acc[1][i], false, false);
  }
}

// ---------------- fused joint: relu(f+g) -> GEMM(W_joint) -> log_softmax ------------
// grid: B*T*(U/32) = 4096 blocks, 256 threads (8 waves).
__global__ __launch_bounds__(256)
void k_joint(const float* __restrict__ f,              // (B*T, H) fp32
             const float* __restrict__ g,              // (B*U, H) fp32
             const unsigned short* __restrict__ Wp,    // packed bf16 (KC=20, NT=64)
             const unsigned short* __restrict__ w1024, // bf16[640]
             const float* __restrict__ bj,             // (1025)
             float* __restrict__ out) {
  __shared__ float gstage[32 * HH];          // 80 KB, staged via TDM / async
  __shared__ float fstage[HH];               // 2.5 KB, staged via async-to-LDS
  __shared__ unsigned short hsh[32 * HP];    // 40.5 KB bf16 activations
  __shared__ float wred[8][32];
  __shared__ float rowmax[32];
  __shared__ float rowlse[32];
  __shared__ float c1024[32];

  const int tid  = threadIdx.x;
  const int lane = tid & 31;
  const int w    = tid >> 5;
  const int blk  = blockIdx.x;
  const int ut   = blk & 3;
  const int bt   = blk >> 2;            // (b*T + t)
  const int b    = bt >> 8;             // T = 256
  const int u0   = ut * 32;

  // ---- stage f row + 32 g rows into LDS (CDNA5 TDM / async-to-LDS) ----
  {
    const float* gbase = g + (long)(b * UU + u0) * HH;
    const float* fbase = f + (long)bt * HH;
#if HAVE_TDM
    if (w == 0) {
      tdm_load_2d_f32((unsigned int)(uintptr_t)&gstage[0], gbase,
                      /*tile_d0=*/HH, /*tile_d1=*/32, /*tensor_d0=*/HH, /*stride0=*/HH);
    }
#else
    for (int c = tid; c < (32 * HH) / 4; c += 256)
      async_copy_b128((unsigned int)(uintptr_t)&gstage[c * 4], gbase + c * 4);
#endif
    if (tid < HH / 4)
      async_copy_b128((unsigned int)(uintptr_t)&fstage[tid * 4], fbase + tid * 4);

    // warm L2 with this wave's whole W_joint stream while staging completes
    #pragma unroll
    for (int kc = 0; kc < 20; ++kc)
      __builtin_prefetch(Wp + (((long)kc * 64 + w * 8) * 32 + lane) * 16, 0, 0);

    wait_async0();
#if HAVE_TDM
#if __has_builtin(__builtin_amdgcn_s_wait_tensorcnt)
    __builtin_amdgcn_s_wait_tensorcnt(0);
#else
    asm volatile("s_wait_tensorcnt 0x0" ::: "memory");
#endif
#endif
    __syncthreads();
  }

  // ---- build h = relu(f+g) once, bf16, into LDS ----
  {
    int row = tid >> 3;                 // 0..31
    int k0  = (tid & 7) * 80;           // 8 threads cover 640
    const float* frow = fstage + k0;
    const float* grow = gstage + row * HH + k0;
    unsigned short* hrow = hsh + row * HP + k0;
    #pragma unroll 4
    for (int k = 0; k < 80; k += 4) {
      float4 fv = *(const float4*)(frow + k);
      float4 gv = *(const float4*)(grow + k);
      ushort4 hv;
      hv.x = f2bf(fmaxf(fv.x + gv.x, 0.f));
      hv.y = f2bf(fmaxf(fv.y + gv.y, 0.f));
      hv.z = f2bf(fmaxf(fv.z + gv.z, 0.f));
      hv.w = f2bf(fmaxf(fv.w + gv.w, 0.f));
      *(ushort4*)(hrow + k) = hv;
    }
  }
  __syncthreads();

  // ---- WMMA GEMM: 2 M-subtiles x 8 N-tiles per wave, double-buffered B tiles ----
  v8f acc[2][8];
  #pragma unroll
  for (int ms = 0; ms < 2; ++ms)
    #pragma unroll
    for (int i = 0; i < 8; ++i)
      #pragma unroll
      for (int r = 0; r < 8; ++r) acc[ms][i][r] = 0.f;

  const int kb = (lane >> 4) << 3;
  const unsigned short* hr0 = hsh + (lane & 15) * HP;
  const unsigned short* hr1 = hsh + ((lane & 15) + 16) * HP;

  {
    FragB b0[8], b1[8];
    load_btiles(Wp, 0, w, lane, b0);
    for (int kc = 0; kc < 20; kc += 2) {
      load_btiles(Wp, kc + 1, w, lane, b1);     // overlap with wmma chain for kc
      wmma_step(hr0, hr1, kc, kb, b0, acc);
      if (kc + 2 < 20) load_btiles(Wp, kc + 2, w, lane, b0); // overlap with kc+1 chain
      wmma_step(hr0, hr1, kc + 1, kb, b1, acc);
    }
  }

  // ---- bias add ----
  #pragma unroll
  for (int i = 0; i < 8; ++i) {
    float bv = bj[w * 128 + i * 16 + (lane & 15)];
    #pragma unroll
    for (int ms = 0; ms < 2; ++ms)
      #pragma unroll
      for (int r = 0; r < 8; ++r) acc[ms][i][r] += bv;
  }

  // ---- column 1024 (blank) via VALU dot ----
  {
    int row = tid >> 3;
    int k0  = (tid & 7) * 80;
    const unsigned short* hrow = hsh + row * HP + k0;
    float s = 0.f;
    for (int k = 0; k < 80; ++k) s += bf2f(hrow[k]) * bf2f(w1024[k0 + k]);
    s += __shfl_xor(s, 1, 32);
    s += __shfl_xor(s, 2, 32);
    s += __shfl_xor(s, 4, 32);
    if ((tid & 7) == 0) c1024[row] = s + bj[1024];
  }
  __syncthreads();

  // ---- row max ----
  #pragma unroll
  for (int ms = 0; ms < 2; ++ms) {
    #pragma unroll
    for (int r = 0; r < 8; ++r) {
      float m = acc[ms][0][r];
      #pragma unroll
      for (int i = 1; i < 8; ++i) m = fmaxf(m, acc[ms][i][r]);
      m = fmaxf(m, __shfl_xor(m, 1, 32));
      m = fmaxf(m, __shfl_xor(m, 2, 32));
      m = fmaxf(m, __shfl_xor(m, 4, 32));
      m = fmaxf(m, __shfl_xor(m, 8, 32));
      int row = ms * 16 + r + ((lane >> 4) << 3);
      if ((lane & 15) == 0) wred[w][row] = m;
    }
  }
  __syncthreads();
  if (tid < 32) {
    float m = c1024[tid];
    #pragma unroll
    for (int ww = 0; ww < 8; ++ww) m = fmaxf(m, wred[ww][tid]);
    rowmax[tid] = m;
  }
  __syncthreads();

  // ---- row sum(exp) -> logsumexp ----
  #pragma unroll
  for (int ms = 0; ms < 2; ++ms) {
    #pragma unroll
    for (int r = 0; r < 8; ++r) {
      int row = ms * 16 + r + ((lane >> 4) << 3);
      float mm = rowmax[row];
      float s = 0.f;
      #pragma unroll
      for (int i = 0; i < 8; ++i) s += __expf(acc[ms][i][r] - mm);
      s += __shfl_xor(s, 1, 32);
      s += __shfl_xor(s, 2, 32);
      s += __shfl_xor(s, 4, 32);
      s += __shfl_xor(s, 8, 32);
      if ((lane & 15) == 0) wred[w][row] = s;
    }
  }
  __syncthreads();
  if (tid < 32) {
    float s = __expf(c1024[tid] - rowmax[tid]);
    #pragma unroll
    for (int ww = 0; ww < 8; ++ww) s += wred[ww][tid];
    rowlse[tid] = rowmax[tid] + __logf(s);
  }
  __syncthreads();

  // ---- store log-softmax ----
  const long obase = ((long)bt * UU + u0) * VV;
  #pragma unroll
  for (int ms = 0; ms < 2; ++ms) {
    #pragma unroll
    for (int r = 0; r < 8; ++r) {
      int row = ms * 16 + r + ((lane >> 4) << 3);
      float lz = rowlse[row];
      long rbase = obase + (long)row * VV + w * 128 + (lane & 15);
      #pragma unroll
      for (int i = 0; i < 8; ++i) out[rbase + i * 16] = acc[ms][i][r] - lz;
    }
  }
  if (tid < 32) out[obase + (long)tid * VV + 1024] = c1024[tid] - rowlse[tid];
}

// ---------------- workspace layout (bytes) ----------------
static const size_t OFF_ENCT   = 0;
static const size_t OFF_DECT   = OFF_ENCT   + 2097152;
static const size_t OFF_WENCP  = OFF_DECT   + 655360;
static const size_t OFF_WPREDP = OFF_WENCP  + 1310720;
static const size_t OFF_WJP    = OFF_WPREDP + 819200;
static const size_t OFF_W1024  = OFF_WJP    + 1310720;
static const size_t OFF_F      = OFF_W1024  + 2048;
static const size_t OFF_G      = OFF_F      + 2621440;

extern "C" void kernel_launch(void* const* d_in, const int* in_sizes, int n_in,
                              void* d_out, int out_size, void* d_ws, size_t ws_size,
                              hipStream_t stream) {
  (void)in_sizes; (void)n_in; (void)out_size; (void)ws_size;
  const float* enc     = (const float*)d_in[0];
  const float* dec     = (const float*)d_in[1];
  const float* W_enc   = (const float*)d_in[2];
  const float* b_enc   = (const float*)d_in[3];
  const float* W_pred  = (const float*)d_in[4];
  const float* b_pred  = (const float*)d_in[5];
  const float* W_joint = (const float*)d_in[6];
  const float* b_joint = (const float*)d_in[7];
  float* out = (float*)d_out;
  char* ws = (char*)d_ws;

  unsigned short* encT   = (unsigned short*)(ws + OFF_ENCT);
  unsigned short* decT   = (unsigned short*)(ws + OFF_DECT);
  unsigned short* wencp  = (unsigned short*)(ws + OFF_WENCP);
  unsigned short* wpredp = (unsigned short*)(ws + OFF_WPREDP);
  unsigned short* wjp    = (unsigned short*)(ws + OFF_WJP);
  unsigned short* w1024  = (unsigned short*)(ws + OFF_W1024);
  float* fbuf = (float*)(ws + OFF_F);
  float* gbuf = (float*)(ws + OFF_G);

  k_transpose_cast<<<(BB*EE*TT)/256, 256, 0, stream>>>(enc, encT, BB, EE, TT);
  k_transpose_cast<<<(BB*PP*UU)/256, 256, 0, stream>>>(dec, decT, BB, PP, UU);
  k_packB<<<(HH*EE)/256, 256, 0, stream>>>(W_enc,  wencp,  HH, EE);
  k_packB<<<(HH*PP)/256, 256, 0, stream>>>(W_pred, wpredp, HH, PP);
  k_packB<<<(VP*HH)/256, 256, 0, stream>>>(W_joint, wjp, VP, HH);
  k_pack_w1024<<<3, 256, 0, stream>>>(W_joint, w1024);

  k_gemm_bias<<<dim3((BB*TT)/16, HH/64), 32, 0, stream>>>(encT, wencp, b_enc, fbuf,
                                                          BB*TT, HH, EE);
  k_gemm_bias<<<dim3((BB*UU)/16, HH/64), 32, 0, stream>>>(decT, wpredp, b_pred, gbuf,
                                                          BB*UU, HH, PP);

  k_joint<<<BB*TT*(UU/32), 256, 0, stream>>>(fbuf, gbuf, wjp, w1024, b_joint, out);
}
